// LSTM_2156073583328
// MI455X (gfx1250) — matrix operand order repaired and checked by reading, not compiled, not run on hardware
//
#include <hip/hip_runtime.h>
#include <hip/hip_bf16.h>

// LSTM (L=2, S=512, B=64, H=1024) for gfx1250 / MI455X.
// bf16 WMMA (16x16x32, f32 accum) for both GEMMs of each cell, fused with
// bias + sigmoid/tanh + state update in one kernel per (timestep, layer).
// Weights converted to bf16 once per launch (33 MB -> L2-resident across the
// 512-step recurrence). Operand staging uses GLOBAL_LOAD_ASYNC_TO_LDS_B128
// (ASYNCcnt) with 2-deep LDS double buffering to hide L2->LDS latency behind
// the WMMAs on the latency-critical serial chain.

typedef __attribute__((ext_vector_type(16))) __bf16 v16bf;
typedef __attribute__((ext_vector_type(8)))  float  v8f;

#define S_LEN 512
#define BATCH 64
#define HID   1024
#define G4    4096
#define NLAY  2
#define KC    32          // K-chunk staged in LDS per pipeline stage

// ---- workspace layout (bytes) ----
static constexpr size_t WXB_OFF = 0;                                          // bf16 Wx [L][4H][H]
static constexpr size_t WXB_SZ  = (size_t)NLAY * G4 * HID * 2;                // 16 MB
static constexpr size_t WHB_OFF = WXB_OFF + WXB_SZ;                           // bf16 Wh
static constexpr size_t WHB_SZ  = WXB_SZ;
static constexpr size_t HB_OFF  = WHB_OFF + WHB_SZ;                           // bf16 h, double buffered: [2][L][B][H]
static constexpr size_t HB_SZ   = (size_t)2 * NLAY * BATCH * HID * 2;
static constexpr size_t HF_OFF  = HB_OFF + HB_SZ;                             // f32 h state [L][B][H]
static constexpr size_t HF_SZ   = (size_t)NLAY * BATCH * HID * 4;
static constexpr size_t CF_OFF  = HF_OFF + HF_SZ;                             // f32 c state [L][B][H]

__device__ __forceinline__ float sigmoidf_(float x) {
  return 1.0f / (1.0f + __expf(-x));
}

// Async global->LDS copy, 16 bytes per lane. VDST = per-lane LDS byte offset,
// VADDR = per-lane 64-bit global address (GV mode). Tracked by ASYNCcnt.
__device__ __forceinline__ void async_b128(unsigned lds_off, const void* gsrc) {
  asm volatile("global_load_async_to_lds_b128 %0, %1, off"
               :: "v"(lds_off), "v"((unsigned long long)gsrc)
               : "memory");
}
__device__ __forceinline__ void wait_async0() {
  asm volatile("s_wait_asynccnt 0x0" ::: "memory");
}
// Generic pointer to an LDS object: low 32 bits are the LDS byte offset.
__device__ __forceinline__ unsigned lds_off_of(const void* p) {
  return (unsigned)(unsigned long long)p;
}

__global__ void cvt_f32_bf16(const float* __restrict__ src,
                             __bf16* __restrict__ dst, int n) {
  int i = blockIdx.x * blockDim.x + threadIdx.x;
  if (i < n) dst[i] = (__bf16)src[i];
}

__global__ void init_state(const float* __restrict__ h0, const float* __restrict__ c0,
                           float* __restrict__ hf, float* __restrict__ cf,
                           __bf16* __restrict__ hb_slot0, int n) {
  int i = blockIdx.x * blockDim.x + threadIdx.x;
  if (i < n) {
    float h = h0[i];
    hf[i] = h;
    hb_slot0[i] = (__bf16)h;
    cf[i] = c0[i];
  }
}

__global__ void finalize(const float* __restrict__ hf, const float* __restrict__ cf,
                         float* __restrict__ out, int n) {
  int i = blockIdx.x * blockDim.x + threadIdx.x;
  if (i < n) {
    out[i]     = hf[i];   // h_fin
    out[n + i] = cf[i];   // c_fin
  }
}

// One LSTM cell: gates[64,4096] = in@Wx^T + h_prev@Wh^T + bx + bh, then
// f,i,o = sigmoid; g = tanh; c = f*c + i*g; h = o*tanh(c).
// Grid: 32 blocks (one per 32-wide H slice), 256 threads = 8 waves.
// Wave w: batch rows m0=(w&3)*16, hcol offset (w>>2)*16; owns one 16x16 tile
// per gate so the gate combine is register-local.
__global__ __launch_bounds__(256) void lstm_step(
    const void*  __restrict__ in,        // [B,H] f32 (layer0) or bf16 (layer1)
    int in_is_f32,
    const __bf16* __restrict__ Wxb,      // layer slice [4H][H]
    const __bf16* __restrict__ Whb,      // layer slice [4H][H]
    const float* __restrict__ bxl,       // layer slice [4H]
    const float* __restrict__ bhl,       // layer slice [4H]
    const __bf16* __restrict__ hb_in,    // [B,H] prev h, bf16 (read slot)
    float* __restrict__ cf,              // [B,H] c state (RMW)
    float* __restrict__ hf,              // [B,H] h state f32 (write)
    __bf16* __restrict__ hb_out,         // [B,H] new h bf16 (write slot)
    float* __restrict__ ys_out)          // nullptr, or ys[t] slice [B,H]
{
  // Double-buffered staging (48 KB total)
  __shared__ __bf16 Ain[2][BATCH][KC];   // input rows          (8 KB)
  __shared__ __bf16 Ah[2][BATCH][KC];    // h_prev rows         (8 KB)
  __shared__ __bf16 Bw[2][2][128][KC];   // [buf][Wx|Wh][gate-rows][k] (32 KB)

  const int tid   = threadIdx.x;
  const int wg    = blockIdx.x;          // h-col block: [wg*32, wg*32+32)
  const int wave  = tid >> 5;
  const int lane  = tid & 31;
  const int m0    = (wave & 3) * 16;     // batch tile
  const int hcoff = (wave >> 2) * 16;    // hcol tile within the 32-slice
  const int lhalf = (lane >> 4) & 1;     // lane half (ISA fragment layout)
  const int l15   = lane & 15;

  v8f acc[4];                            // f, i, o, c_ex accumulators
  {
    v8f z = {0.f, 0.f, 0.f, 0.f, 0.f, 0.f, 0.f, 0.f};
    acc[0] = z; acc[1] = z; acc[2] = z; acc[3] = z;
  }

  // Issue all staging traffic for one K-chunk into LDS buffer `buf`.
  auto issue_stage = [&](int buf, int kc) {
    // h_prev rows: 64 x 32 bf16 = 256 x 16B chunks -> 1 async per thread
    for (int j = tid; j < BATCH * (KC / 8); j += 256) {
      int r = j >> 2, c8 = (j & 3) * 8;
      async_b128(lds_off_of(&Ah[buf][r][c8]),
                 hb_in + (size_t)r * HID + kc + c8);
    }
    if (in_is_f32) {
      // layer 0: convert f32 x -> bf16 through VGPRs (needs VALU anyway)
      const float* inf = (const float*)in;
      for (int j = tid; j < BATCH * KC; j += 256) {
        int r = j >> 5, c = j & (KC - 1);
        Ain[buf][r][c] = (__bf16)inf[(size_t)r * HID + kc + c];
      }
    } else {
      const __bf16* inb = (const __bf16*)in;
      for (int j = tid; j < BATCH * (KC / 8); j += 256) {
        int r = j >> 2, c8 = (j & 3) * 8;
        async_b128(lds_off_of(&Ain[buf][r][c8]),
                   inb + (size_t)r * HID + kc + c8);
      }
    }
    // weights: 2 matrices x 128 gate-rows x 32 k = 1024 chunks -> 4 per thread
    for (int j = tid; j < 2 * 128 * (KC / 8); j += 256) {
      int mtx = j >> 9;                  // 0 = Wx, 1 = Wh
      int r   = (j >> 2) & 127;          // gate-local row
      int c8  = (j & 3) * 8;             // 8 bf16 = 16 bytes
      int g   = r >> 5;
      int rr  = r & 31;
      const __bf16* src = mtx ? Whb : Wxb;
      async_b128(lds_off_of(&Bw[buf][mtx][r][c8]),
                 src + (size_t)(g * HID + wg * 32 + rr) * HID + kc + c8);
    }
  };

  // ---- software pipeline: prologue issue, then wait/publish -> prefetch -> compute
  issue_stage(0, 0);
  const int NCHUNK = HID / KC;           // 32
  for (int i = 0; i < NCHUNK; ++i) {
    const int c = i & 1;
    wait_async0();                       // own wave's asyncs for chunk i done
    __syncthreads();                     // publish buf c; all waves done reading buf c^1
    if (i + 1 < NCHUNK) issue_stage(c ^ 1, (i + 1) * KC);

    // A fragment (16x32 bf16): lanes 0-15 hold K{0..7,16..23}, 16-31 K{8..15,24..31}
    union { uint4 u[2]; v16bf v; } fa_in, fa_h;
    const int ak = lhalf ? 8 : 0;
    fa_in.u[0] = *(const uint4*)&Ain[c][m0 + l15][ak];
    fa_in.u[1] = *(const uint4*)&Ain[c][m0 + l15][ak + 16];
    fa_h.u[0]  = *(const uint4*)&Ah[c][m0 + l15][ak];
    fa_h.u[1]  = *(const uint4*)&Ah[c][m0 + l15][ak + 16];
    // B fragment (32x16 bf16): lane = N col, lanes 0-15 hold K{0..7,16..23}, 16-31 K{8..15,24..31}
    const int bk = lhalf ? 8 : 0;
    #pragma unroll
    for (int g = 0; g < 4; ++g) {
      union { uint4 u[2]; v16bf v; } fb_x, fb_h;
      const int br = g * 32 + hcoff + l15;
      fb_x.u[0] = *(const uint4*)&Bw[c][0][br][bk];
      fb_x.u[1] = *(const uint4*)&Bw[c][0][br][bk + 16];
      fb_h.u[0] = *(const uint4*)&Bw[c][1][br][bk];
      fb_h.u[1] = *(const uint4*)&Bw[c][1][br][bk + 16];
      acc[g] = __builtin_amdgcn_wmma_f32_16x16x32_bf16(
          false, fa_in.v, false, fb_x.v, (short)0, acc[g], false, false);
      acc[g] = __builtin_amdgcn_wmma_f32_16x16x32_bf16(
          false, fa_h.v,  false, fb_h.v, (short)0, acc[g], false, false);
    }
  }

  // ---- epilogue: bias + activations + state update (register-local) ----
  const int hcol = wg * 32 + hcoff + l15;          // this lane's N column
  float bsum[4];
  #pragma unroll
  for (int g = 0; g < 4; ++g)
    bsum[g] = bxl[g * HID + hcol] + bhl[g * HID + hcol];

  #pragma unroll
  for (int r = 0; r < 8; ++r) {
    // C/D layout: VGPR r -> M=r (lanes 0-15) / M=8+r (lanes 16-31)
    const int m = m0 + r + (lhalf ? 8 : 0);
    const size_t idx = (size_t)m * HID + hcol;
    float gf = sigmoidf_(acc[0][r] + bsum[0]);
    float gi = sigmoidf_(acc[1][r] + bsum[1]);
    float go = sigmoidf_(acc[2][r] + bsum[2]);
    float gc = tanhf(acc[3][r] + bsum[3]);
    float cn = gf * cf[idx] + gi * gc;
    float hn = go * tanhf(cn);
    cf[idx] = cn;
    hf[idx] = hn;
    hb_out[idx] = (__bf16)hn;
    if (ys_out) ys_out[idx] = hn;
  }
}

extern "C" void kernel_launch(void* const* d_in, const int* in_sizes, int n_in,
                              void* d_out, int out_size, void* d_ws, size_t ws_size,
                              hipStream_t stream) {
  const float* x  = (const float*)d_in[0];   // [S,B,H]
  const float* h0 = (const float*)d_in[1];   // [L,B,H]
  const float* c0 = (const float*)d_in[2];   // [L,B,H]
  const float* Wx = (const float*)d_in[3];   // [L,4H,H]
  const float* Wh = (const float*)d_in[4];   // [L,4H,H]
  const float* bx = (const float*)d_in[5];   // [L,4H]
  const float* bh = (const float*)d_in[6];   // [L,4H]
  float* out = (float*)d_out;                // ys | h_fin | c_fin

  char* ws = (char*)d_ws;
  __bf16* Wxb = (__bf16*)(ws + WXB_OFF);
  __bf16* Whb = (__bf16*)(ws + WHB_OFF);
  __bf16* HB  = (__bf16*)(ws + HB_OFF);      // [2][L][B][H]
  float*  HF  = (float*)(ws + HF_OFF);       // [L][B][H]
  float*  CF  = (float*)(ws + CF_OFF);       // [L][B][H]

  const int nw = NLAY * G4 * HID;
  cvt_f32_bf16<<<(nw + 255) / 256, 256, 0, stream>>>(Wx, Wxb, nw);
  cvt_f32_bf16<<<(nw + 255) / 256, 256, 0, stream>>>(Wh, Whb, nw);

  const int ns = NLAY * BATCH * HID;
  init_state<<<(ns + 255) / 256, 256, 0, stream>>>(h0, c0, HF, CF, HB, ns);

  const size_t BH     = (size_t)BATCH * HID;
  const size_t layerW = (size_t)G4 * HID;

  for (int t = 0; t < S_LEN; ++t) {
    const int rs = t & 1;            // read slot
    const int wsl = (t + 1) & 1;     // write slot
    // layer 0: input = x[t] (f32)
    lstm_step<<<32, 256, 0, stream>>>(
        x + (size_t)t * BH, 1,
        Wxb, Whb, bx, bh,
        HB + ((size_t)rs * NLAY + 0) * BH,
        CF, HF,
        HB + ((size_t)wsl * NLAY + 0) * BH,
        nullptr);
    // layer 1: input = layer0's new h (bf16, write slot), output -> ys[t]
    lstm_step<<<32, 256, 0, stream>>>(
        HB + ((size_t)wsl * NLAY + 0) * BH, 0,
        Wxb + layerW, Whb + layerW, bx + G4, bh + G4,
        HB + ((size_t)rs * NLAY + 1) * BH,
        CF + BH, HF + BH,
        HB + ((size_t)wsl * NLAY + 1) * BH,
        out + (size_t)t * BH);
  }

  finalize<<<(ns + 255) / 256, 256, 0, stream>>>(HF, CF, out + (size_t)S_LEN * BH, ns);
}
